// GridGAT_79766132621695
// MI455X (gfx1250) — matrix operand order repaired and checked by reading, not compiled, not run on hardware
//
#include <hip/hip_runtime.h>
#include <math.h>

typedef __attribute__((ext_vector_type(16))) _Float16 v16h;
typedef __attribute__((ext_vector_type(8)))  float    v8f;

#define NEG_SLOPE 0.2f
#define HC 512      // heads * out_c for layer 1
#define H1 8
#define C1 64

// ---------- native FP32 atomic add (CDNA5 has GLOBAL_ATOMIC_ADD_F32)
static __device__ __forceinline__ void atomic_add_f32(float* p, float v) {
    unsafeAtomicAdd(p, v);
}

// ---------- order-preserving float<->uint map for atomicMax-based segment max
static __device__ __forceinline__ unsigned order_f32(float f) {
    unsigned b = __float_as_uint(f);
    return (b & 0x80000000u) ? ~b : (b | 0x80000000u);
}
static __device__ __forceinline__ float unorder_f32(unsigned k) {
    unsigned b = (k & 0x80000000u) ? (k ^ 0x80000000u) : ~k;
    return __uint_as_float(b);
}

// ---------- Layer-1 GEMM: h1[N,512] = x[N,11] @ W1[11,512] via WMMA f16->f32
// One wave computes FOUR adjacent 16x16 output tiles (same 16 rows), loading
// the A fragment once and issuing 4 v_wmma_f32_16x16x32_f16 (K=11 padded to 32).
// All fragment loads are unconditional (clamped addresses); K-masking is done
// on the loaded VALUE (v_cndmask) so no exec-mask branches are generated.
__global__ void gat_gemm1_wmma(const float* __restrict__ x,
                               const float* __restrict__ W1,
                               float* __restrict__ h1,
                               int mtiles /* N/16 */) {
    const int lane = threadIdx.x & 31;
    const int wave = threadIdx.x >> 5;
    const int wt = blockIdx.x * 4 + wave;       // wave-tile id
    const int nwt = mtiles * 8;                 // 8 col-groups of 4 tiles each
    if (wt >= nwt) return;
    const int rt = wt >> 3;                     // row tile
    const int cg = wt & 7;                      // col group (4 tiles)
    const int lr = lane & 15;
    const int g  = lane >> 4;

    const float* __restrict__ xr = x + (long long)(rt * 16 + lr) * 11;

    // A 16x32 f16 layout (contiguous, matches B): element e -> K = g*16+e.
    // Only g==0, e<=10 are nonzero (K < 11).
    v16h a;
#pragma unroll
    for (int e = 0; e < 16; ++e) a[e] = (_Float16)0.0f;
#pragma unroll
    for (int e = 0; e < 11; ++e) {
        const int ka = g * 16 + e;
        const float va = xr[ka < 11 ? ka : 0];          // unconditional load
        a[e] = (_Float16)(ka < 11 ? va : 0.0f);         // value select (cndmask)
    }

#pragma unroll
    for (int q = 0; q < 4; ++q) {
        const int ct = cg * 4 + q;
        // B 32x16 f16 layout: element e of lane L -> K = (L/16)*16 + e, N = L%16.
        // Valid only for g==0, e<=10; address is g-independent there.
        v16h b;
#pragma unroll
        for (int e = 0; e < 16; ++e) b[e] = (_Float16)0.0f;
#pragma unroll
        for (int e = 0; e < 11; ++e) {
            const float vb = W1[e * HC + ct * 16 + lr]; // unconditional, coalesced
            b[e] = (_Float16)(g == 0 ? vb : 0.0f);      // value select (cndmask)
        }
        v8f c = {};
        c = __builtin_amdgcn_wmma_f32_16x16x32_f16(false, a, false, b, (short)0, c, false, false);
#pragma unroll
        for (int r = 0; r < 8; ++r) {
            const int m = r + g * 8;   // C/D layout: lanes 0-15 -> M=r, lanes 16-31 -> M=r+8
            h1[(long long)(rt * 16 + m) * HC + ct * 16 + lr] = c[r];
        }
    }
}

// scalar tail for rows not covered by 16-row tiles (not used for N=50000)
__global__ void gat_gemm1_tail(const float* __restrict__ x,
                               const float* __restrict__ W1,
                               float* __restrict__ h1,
                               int row0, int N) {
    const int t = blockIdx.x * blockDim.x + threadIdx.x;
    const int rows = N - row0;
    if (t >= rows * HC) return;
    const int r = row0 + t / HC, c = t % HC;
    float acc = 0.0f;
#pragma unroll
    for (int k = 0; k < 11; ++k) acc += x[(long long)r * 11 + k] * W1[k * HC + c];
    h1[(long long)r * HC + c] = acc;
}

// ---------- per-(node,head) attention coefficients: a = <h[n,hh,:], att[hh,:]>
__global__ void gat_attn_coef1(const float* __restrict__ h1,
                               const float* __restrict__ att_src,
                               const float* __restrict__ att_dst,
                               float* __restrict__ a_src,
                               float* __restrict__ a_dst, int N) {
    const int w = (blockIdx.x * blockDim.x + threadIdx.x) >> 5;
    const int lane = threadIdx.x & 31;
    if (w >= N * H1) return;
    const int n = w >> 3, hh = w & 7;
    const float* __restrict__ hp = h1 + (long long)n * HC + hh * C1;
    const float* __restrict__ as = att_src + hh * C1;
    const float* __restrict__ ad = att_dst + hh * C1;
    float ss = hp[lane] * as[lane] + hp[lane + 32] * as[lane + 32];
    float sd = hp[lane] * ad[lane] + hp[lane + 32] * ad[lane + 32];
#pragma unroll
    for (int off = 16; off; off >>= 1) {
        ss += __shfl_xor(ss, off, 32);
        sd += __shfl_xor(sd, off, 32);
    }
    if (lane == 0) { a_src[w] = ss; a_dst[w] = sd; }
}

// ---------- edge pass 1: segment max of leaky_relu(a_src[src]+a_dst[dst])
__global__ void gat_edge_max(const int* __restrict__ ei, int E, int Etot,
                             const float* __restrict__ a_src,
                             const float* __restrict__ a_dst,
                             unsigned* __restrict__ m, int H) {
    const long long t = (long long)blockIdx.x * blockDim.x + threadIdx.x;
    if (t >= (long long)Etot * H) return;
    const int e = (int)(t / H);
    const int hh = (int)(t - (long long)e * H);
    int s, d;
    if (e < E) { s = ei[e]; d = ei[E + e]; } else { s = d = e - E; }
    float v = a_src[(long long)s * H + hh] + a_dst[(long long)d * H + hh];
    v = v > 0.0f ? v : NEG_SLOPE * v;
    atomicMax(&m[(long long)d * H + hh], order_f32(v));
}

// ---------- edge pass 2: w = exp(e - m[dst]); s[dst] += w
__global__ void gat_edge_expsum(const int* __restrict__ ei, int E, int Etot,
                                const float* __restrict__ a_src,
                                const float* __restrict__ a_dst,
                                const unsigned* __restrict__ m,
                                float* __restrict__ w,
                                float* __restrict__ ssum, int H) {
    const long long t = (long long)blockIdx.x * blockDim.x + threadIdx.x;
    if (t >= (long long)Etot * H) return;
    const int e = (int)(t / H);
    const int hh = (int)(t - (long long)e * H);
    int s, d;
    if (e < E) { s = ei[e]; d = ei[E + e]; } else { s = d = e - E; }
    float v = a_src[(long long)s * H + hh] + a_dst[(long long)d * H + hh];
    v = v > 0.0f ? v : NEG_SLOPE * v;
    const float mv = unorder_f32(m[(long long)d * H + hh]);
    const float wv = __expf(v - mv);
    w[(long long)e * H + hh] = wv;
    atomic_add_f32(&ssum[(long long)d * H + hh], wv);
}

// ---------- edge pass 3 (layer 1): out[dst, ch] += h1[src, ch] * alpha[head(ch)]
// one wave per edge; float4 gathers of h1[src], 16 HW f32 atomics per lane
__global__ void gat_edge_agg1(const int* __restrict__ ei, int E, int Etot,
                              const float* __restrict__ h1,
                              const float* __restrict__ w,
                              const float* __restrict__ ssum,
                              float* __restrict__ out) {
    const int e = (int)((long long)(blockIdx.x * blockDim.x + threadIdx.x) >> 5);
    const int lane = threadIdx.x & 31;
    if (e >= Etot) return;
    int s, d;
    if (e < E) { s = ei[e]; d = ei[E + e]; } else { s = d = e - E; }
    const float4* __restrict__ hs = (const float4*)(h1 + (long long)s * HC);
    float* __restrict__ od = out + (long long)d * HC;
    const float* __restrict__ we = w + (long long)e * H1;
    const float* __restrict__ sd = ssum + (long long)d * H1;
    float alpha[H1];
#pragma unroll
    for (int hh = 0; hh < H1; ++hh) alpha[hh] = we[hh] / (sd[hh] + 1e-16f);
#pragma unroll
    for (int i = 0; i < 4; ++i) {
        const int v4 = lane + 32 * i;      // float4 index, 0..127
        const float4 hv = hs[v4];
        const int ch = v4 * 4;             // 4 consecutive channels share one head
        const float al = alpha[ch >> 6];
        atomic_add_f32(&od[ch + 0], hv.x * al);
        atomic_add_f32(&od[ch + 1], hv.y * al);
        atomic_add_f32(&od[ch + 2], hv.z * al);
        atomic_add_f32(&od[ch + 3], hv.w * al);
    }
}

// ---------- bias + ELU (eval-mode dropout == identity), float4
__global__ void gat_bias_elu(const float4* __restrict__ in,
                             const float* __restrict__ b1,
                             float4* __restrict__ out, long long total4) {
    const long long t = (long long)blockIdx.x * blockDim.x + threadIdx.x;
    if (t >= total4) return;
    const int cb = (int)(t & (HC / 4 - 1)) * 4;
    float4 v = in[t];
    v.x += b1[cb + 0]; v.y += b1[cb + 1]; v.z += b1[cb + 2]; v.w += b1[cb + 3];
    v.x = v.x > 0.0f ? v.x : (__expf(v.x) - 1.0f);
    v.y = v.y > 0.0f ? v.y : (__expf(v.y) - 1.0f);
    v.z = v.z > 0.0f ? v.z : (__expf(v.z) - 1.0f);
    v.w = v.w > 0.0f ? v.w : (__expf(v.w) - 1.0f);
    out[t] = v;
}

// ---------- layer-2 GEMV + attention coefficients: z = h2 @ W2 (512 -> 1)
__global__ void gat_gemv2(const float* __restrict__ h2,
                          const float* __restrict__ W2,
                          const float* __restrict__ att_s2,
                          const float* __restrict__ att_d2,
                          float* __restrict__ z,
                          float* __restrict__ as2,
                          float* __restrict__ ad2, int N) {
    const int n = (blockIdx.x * blockDim.x + threadIdx.x) >> 5;
    const int lane = threadIdx.x & 31;
    if (n >= N) return;
    const float4* __restrict__ hp = (const float4*)(h2 + (long long)n * HC);
    const float4* __restrict__ wp = (const float4*)W2;
    float sum = 0.0f;
#pragma unroll
    for (int i = 0; i < 4; ++i) {
        const float4 hv = hp[lane + 32 * i];
        const float4 wv = wp[lane + 32 * i];
        sum += hv.x * wv.x + hv.y * wv.y + hv.z * wv.z + hv.w * wv.w;
    }
#pragma unroll
    for (int off = 16; off; off >>= 1) sum += __shfl_xor(sum, off, 32);
    if (lane == 0) {
        z[n] = sum;
        as2[n] = sum * att_s2[0];
        ad2[n] = sum * att_d2[0];
    }
}

// ---------- edge pass 3 (layer 2, H=1, C=1)
__global__ void gat_edge_agg2(const int* __restrict__ ei, int E, int Etot,
                              const float* __restrict__ z,
                              const float* __restrict__ w,
                              const float* __restrict__ ssum,
                              float* __restrict__ out2) {
    const int e = blockIdx.x * blockDim.x + threadIdx.x;
    if (e >= Etot) return;
    int s, d;
    if (e < E) { s = ei[e]; d = ei[E + e]; } else { s = d = e - E; }
    const float alpha = w[e] / (ssum[d] + 1e-16f);
    atomic_add_f32(&out2[d], z[s] * alpha);
}

// ---------- final bias
__global__ void gat_final(const float* __restrict__ out2,
                          const float* __restrict__ b2,
                          float* __restrict__ dout, int N) {
    const int n = blockIdx.x * blockDim.x + threadIdx.x;
    if (n < N) dout[n] = out2[n] + b2[0];
}

extern "C" void kernel_launch(void* const* d_in, const int* in_sizes, int n_in,
                              void* d_out, int out_size, void* d_ws, size_t ws_size,
                              hipStream_t stream) {
    const float* x        = (const float*)d_in[0];
    const int*   ei       = (const int*)d_in[1];   // [2,E] src row then dst row
    const float* W1       = (const float*)d_in[2];
    const float* att_src1 = (const float*)d_in[3];
    const float* att_dst1 = (const float*)d_in[4];
    const float* b1       = (const float*)d_in[5];
    const float* W2       = (const float*)d_in[6];
    const float* att_s2   = (const float*)d_in[7];
    const float* att_d2   = (const float*)d_in[8];
    const float* b2       = (const float*)d_in[9];

    const int N = in_sizes[0] / 11;
    const int E = in_sizes[1] / 2;
    const int Etot = E + N;                         // with self loops

    // ---- workspace layout (byte cursor, 256B aligned slices)
    char* cur = (char*)d_ws;
    auto take = [&](size_t bytes) {
        void* p = (void*)cur;
        cur += (bytes + 255) & ~(size_t)255;
        return p;
    };
    float*    h1   = (float*)take((size_t)N * HC * 4);   // layer-1 features; reused for h2
    float*    out1 = (float*)take((size_t)N * HC * 4);   // layer-1 aggregate
    float*    as1  = (float*)take((size_t)N * H1 * 4);
    float*    ad1  = (float*)take((size_t)N * H1 * 4);
    unsigned* m1   = (unsigned*)take((size_t)N * H1 * 4);
    float*    s1   = (float*)take((size_t)N * H1 * 4);
    float*    w1e  = (float*)take((size_t)Etot * H1 * 4);
    float*    z    = (float*)take((size_t)N * 4);
    float*    as2  = (float*)take((size_t)N * 4);
    float*    ad2  = (float*)take((size_t)N * 4);
    unsigned* m2   = (unsigned*)take((size_t)N * 4);
    float*    s2   = (float*)take((size_t)N * 4);
    float*    w2e  = (float*)take((size_t)Etot * 4);
    float*    o2   = (float*)take((size_t)N * 4);

    // ---- zero-init accumulators (memset 0 == ordered -inf for m buffers)
    hipMemsetAsync(out1, 0, (size_t)N * HC * 4, stream);
    hipMemsetAsync(m1,   0, (size_t)N * H1 * 4, stream);
    hipMemsetAsync(s1,   0, (size_t)N * H1 * 4, stream);
    hipMemsetAsync(m2,   0, (size_t)N * 4, stream);
    hipMemsetAsync(s2,   0, (size_t)N * 4, stream);
    hipMemsetAsync(o2,   0, (size_t)N * 4, stream);

    auto cdiv = [](long long a, long long b) { return (int)((a + b - 1) / b); };

    // ---- layer-1 GEMM (WMMA): 4 col-tiles per wave
    const int mtiles = N / 16;
    const int nwt = mtiles * 8;
    gat_gemm1_wmma<<<cdiv(nwt, 4), 128, 0, stream>>>(x, W1, h1, mtiles);
    const int rem = N - mtiles * 16;
    if (rem > 0)
        gat_gemm1_tail<<<cdiv((long long)rem * HC, 256), 256, 0, stream>>>(x, W1, h1, mtiles * 16, N);

    // ---- layer-1 attention coefficients
    gat_attn_coef1<<<cdiv((long long)N * H1 * 32, 256), 256, 0, stream>>>(h1, att_src1, att_dst1, as1, ad1, N);

    // ---- layer-1 segment softmax + aggregate
    gat_edge_max<<<cdiv((long long)Etot * H1, 256), 256, 0, stream>>>(ei, E, Etot, as1, ad1, m1, H1);
    gat_edge_expsum<<<cdiv((long long)Etot * H1, 256), 256, 0, stream>>>(ei, E, Etot, as1, ad1, m1, w1e, s1, H1);
    gat_edge_agg1<<<cdiv((long long)Etot * 32, 256), 256, 0, stream>>>(ei, E, Etot, h1, w1e, s1, out1);

    // ---- bias + ELU into h2 (reuse h1 buffer)
    gat_bias_elu<<<cdiv((long long)N * HC / 4, 256), 256, 0, stream>>>(
        (const float4*)out1, b1, (float4*)h1, (long long)N * HC / 4);

    // ---- layer-2 GEMV + coefficients
    gat_gemv2<<<cdiv((long long)N * 32, 256), 256, 0, stream>>>(h1, W2, att_s2, att_d2, z, as2, ad2, N);

    // ---- layer-2 segment softmax + aggregate (H = 1)
    gat_edge_max<<<cdiv(Etot, 256), 256, 0, stream>>>(ei, E, Etot, as2, ad2, m2, 1);
    gat_edge_expsum<<<cdiv(Etot, 256), 256, 0, stream>>>(ei, E, Etot, as2, ad2, m2, w2e, s2, 1);
    gat_edge_agg2<<<cdiv(Etot, 256), 256, 0, stream>>>(ei, E, Etot, z, w2e, s2, o2);

    // ---- final bias -> d_out [N,1]
    gat_final<<<cdiv(N, 256), 256, 0, stream>>>(o2, b2, (float*)d_out, N);
}
